// MoeLayer_78786880078381
// MI455X (gfx1250) — compile-verified, matrix operand order checked by reading, not run on hardware
//
#include <hip/hip_runtime.h>
#include <hip/hip_bf16.h>
#include <math.h>

// Problem dims (fixed by the reference): B=2,S=2048,D=1024,H=4096,E=8
#define TOK   4096              // B*S tokens
#define DDIM  1024
#define HDIM  4096
#define EXP   8
#define OUT_ELEMS (TOK * DDIM)  // 4,194,304 f32 output elements (then TOK*EXP weights)

typedef __attribute__((ext_vector_type(16))) __bf16 bf16x16;
typedef __attribute__((ext_vector_type(2)))  __bf16 bf16x2;
typedef __attribute__((ext_vector_type(8)))  float  f32x8;
typedef __attribute__((ext_vector_type(2)))  float  f32x2;

// ---------- helpers ----------
// Native f32->bf16 (RNE fptrunc; lowers to v_cvt_pk_bf16_f32 when available)
__device__ __forceinline__ unsigned int pk2bf(float lo, float hi) {
    f32x2 s; s.x = lo; s.y = hi;
    union { bf16x2 h; unsigned int u; } cv;
    cv.h = __builtin_convertvector(s, bf16x2);
    return cv.u;
}
__device__ __forceinline__ unsigned short f2bf(float f) {
    union { __bf16 h; unsigned short u; } cv;
    cv.h = (__bf16)f;
    return cv.u;
}
__device__ __forceinline__ float bf2f(unsigned short h) {
    return __uint_as_float(((unsigned int)h) << 16);
}
__device__ __forceinline__ float gelu_exact(float x) {
    return 0.5f * x * (1.0f + erff(x * 0.70710678118654752f));
}
// Wait for all outstanding async (memory<->LDS) copies of this wave.
__device__ __forceinline__ void wait_async0() {
#if __has_builtin(__builtin_amdgcn_s_wait_asynccnt)
    __builtin_amdgcn_s_wait_asynccnt(0);
#else
    asm volatile("s_wait_asynccnt 0" ::: "memory");
#endif
}

// ---------- fp32 -> bf16 bulk convert (x only; weights converted on the fly) ----------
__global__ void __launch_bounds__(256)
f32_to_bf16_kernel(const float* __restrict__ in, unsigned short* __restrict__ out, int n8) {
    int i = blockIdx.x * 256 + threadIdx.x;
    if (i >= n8) return;
    const float4* p = (const float4*)in + (size_t)i * 2;
    float4 a = p[0], b = p[1];
    uint4 o;
    o.x = pk2bf(a.x, a.y); o.y = pk2bf(a.z, a.w);
    o.z = pk2bf(b.x, b.y); o.w = pk2bf(b.z, b.w);
    ((uint4*)out)[i] = o;
}

// ---------- WMMA GEMM: C[M,N] = A_bf16[M,K] * W_f32[N,K]^T (+bias, epilogue) ----------
// MODE 0: out = bf16( gelu(acc + bias[n]) )           (activation chain)
// MODE 1: out_f32[m,n] += gate[m*8] * (acc + bias[n]) (gated expert accumulation)
//
// Pipelined: A tile prefetched k+1 ahead via CDNA5 async memory->LDS DMA
// (global_load_async_to_lds_b128, ASYNCcnt), W tile (f32) prefetched into
// registers before compute and converted/committed to LDS after compute, so
// global latency overlaps the 8 WMMAs of each K-step. Double-buffered LDS.
#define BM 128
#define BN 128
#define BK 32
#define LDSK 40   // 32 + 8 pad: stride 80B keeps b128 frag loads bank-conflict-free & 16B aligned

template <int MODE>
__global__ void __launch_bounds__(256)
gemm_bf16_kernel(const unsigned short* __restrict__ A,   // [M][K] bf16, K-major
                 const float* __restrict__ W,            // [N][K] f32,  K-major
                 const float* __restrict__ bias,         // [N]
                 void* __restrict__ outv,
                 const float* __restrict__ gate,         // MODE 1: weights + e, stride EXP
                 int M, int N, int K)
{
    __shared__ unsigned short sA[2][BM * LDSK];   // 2 x 10 KB
    __shared__ unsigned short sB[2][BN * LDSK];   // 2 x 10 KB

    const int tid  = threadIdx.x;
    const int lane = tid & 31;
    const int wave = tid >> 5;         // 8 waves
    const int wm   = wave >> 2;        // 0..1  -> 64-row slab
    const int wn   = wave & 3;         // 0..3  -> 32-col slab
    const int m0   = blockIdx.y * BM;
    const int n0   = blockIdx.x * BN;

    const int ldRow = tid >> 1;        // 0..127
    const int ldSeg = (tid & 1) * 16;  // 16-element K segment

    const unsigned short* gA = A + (size_t)(m0 + ldRow) * K + ldSeg;
    const float*          gW = W + (size_t)(n0 + ldRow) * K + ldSeg;

    f32x8 acc[4][2];
    #pragma unroll
    for (int i = 0; i < 4; ++i)
        #pragma unroll
        for (int j = 0; j < 2; ++j)
            #pragma unroll
            for (int r = 0; r < 8; ++r) acc[i][j][r] = 0.0f;

    const int kb   = (lane >> 4) << 3;   // 0 (lanes 0-15) or 8 (lanes 16-31)
    const int rsel = lane & 15;
    const int nt   = K / BK;

    float4 w0, w1, w2, w3;               // staged W tile (16 f32 per thread)

    auto fetchW = [&](int kt) {          // issue global f32 loads (no wait here)
        const float4* p = (const float4*)(gW + (size_t)kt * BK);
        w0 = p[0]; w1 = p[1]; w2 = p[2]; w3 = p[3];
    };
    auto commitW = [&](int buf) {        // convert + store staged W tile to LDS
        uint4 o0, o1;
        o0.x = pk2bf(w0.x, w0.y); o0.y = pk2bf(w0.z, w0.w);
        o0.z = pk2bf(w1.x, w1.y); o0.w = pk2bf(w1.z, w1.w);
        o1.x = pk2bf(w2.x, w2.y); o1.y = pk2bf(w2.z, w2.w);
        o1.z = pk2bf(w3.x, w3.y); o1.w = pk2bf(w3.z, w3.w);
        uint4* d = (uint4*)&sB[buf][ldRow * LDSK + ldSeg];
        d[0] = o0; d[1] = o1;
    };
    auto asyncA = [&](int kt, int buf) { // CDNA5 async DMA: global bf16 -> LDS, no VGPR staging
        unsigned long long src = (unsigned long long)(gA + (size_t)kt * BK);
        unsigned int dst = (unsigned int)(unsigned long long)&sA[buf][ldRow * LDSK + ldSeg];
        asm volatile("global_load_async_to_lds_b128 %0, %1, off"
                     :: "v"(dst), "v"(src) : "memory");
        asm volatile("global_load_async_to_lds_b128 %0, %1, off offset:16"
                     :: "v"(dst), "v"(src) : "memory");
    };
    auto compute = [&](int buf) {
        // Fragment loads follow the 16-bit A/B VGPR layout:
        // lanes 0-15: K = {0..7, 16..23}; lanes 16-31: K = {8..15, 24..31}
        union Fr { bf16x16 v; uint4 u[2]; };
        Fr fa[4], fb[2];
        #pragma unroll
        for (int mi = 0; mi < 4; ++mi) {
            int row = wm * 64 + mi * 16 + rsel;
            fa[mi].u[0] = *(const uint4*)&sA[buf][row * LDSK + kb];
            fa[mi].u[1] = *(const uint4*)&sA[buf][row * LDSK + kb + 16];
        }
        #pragma unroll
        for (int ni = 0; ni < 2; ++ni) {
            int row = wn * 32 + ni * 16 + rsel;
            fb[ni].u[0] = *(const uint4*)&sB[buf][row * LDSK + kb];
            fb[ni].u[1] = *(const uint4*)&sB[buf][row * LDSK + kb + 16];
        }
        #pragma unroll
        for (int mi = 0; mi < 4; ++mi)
            #pragma unroll
            for (int ni = 0; ni < 2; ++ni)
                acc[mi][ni] = __builtin_amdgcn_wmma_f32_16x16x32_bf16(
                    false, fa[mi].v, false, fb[ni].v,
                    (short)0, acc[mi][ni], false, false);
    };

    // ---- prologue: tile 0 into buffer 0 ----
    fetchW(0);
    asyncA(0, 0);
    commitW(0);          // compiler inserts the loadcnt wait for w0..w3 here
    wait_async0();
    __syncthreads();

    // ---- pipelined main loop ----
    for (int kt = 0; kt < nt; ++kt) {
        const int cur = kt & 1;
        const int nxt = cur ^ 1;
        if (kt + 1 < nt) {
            fetchW(kt + 1);      // global f32 loads in flight across compute
            asyncA(kt + 1, nxt); // async DMA in flight across compute
        }
        compute(cur);
        __syncthreads();         // all waves done reading LDS (incl. buffer `nxt`)
        if (kt + 1 < nt) {
            commitW(nxt);
            wait_async0();       // own async copies done; barrier covers the rest
            __syncthreads();
        }
    }

    // ---- epilogue. C/D layout: VGPR r -> M = r + 8*(lane>=16); N = lane&15. ----
    const int mhalf = (lane >> 4) << 3;
    #pragma unroll
    for (int mi = 0; mi < 4; ++mi) {
        #pragma unroll
        for (int ni = 0; ni < 2; ++ni) {
            const int col = n0 + wn * 32 + ni * 16 + rsel;
            const float bn = bias[col];
            #pragma unroll
            for (int r = 0; r < 8; ++r) {
                const int m = m0 + wm * 64 + mi * 16 + mhalf + r;
                float v = acc[mi][ni][r] + bn;
                if (MODE == 0) {
                    ((unsigned short*)outv)[(size_t)m * N + col] = f2bf(gelu_exact(v));
                } else {
                    float* o = (float*)outv;
                    o[(size_t)m * N + col] += gate[(size_t)m * EXP] * v;
                }
            }
        }
    }
}

// ---------- gate logits + softmax ----------
// gW3 (8x4096 f32 = 128 KB) is staged once per block into LDS (CDNA5: 320 KB/WGP,
// via dynamic shared memory), then 8 waves x 8 tokens reduce against it.
// Cuts gW3 L2 traffic from 512 MB (per-token re-reads) to 8 MB.
__global__ void __launch_bounds__(256)
gate_softmax_kernel(const unsigned short* __restrict__ g2,  // [TOK][HDIM] bf16
                    const float* __restrict__ W3,           // [EXP][HDIM]
                    const float* __restrict__ b3,           // [EXP]
                    float* __restrict__ wout)               // [TOK][EXP]
{
    extern __shared__ float sW3[];            // EXP*HDIM floats = 128 KB
    const int tid  = threadIdx.x;
    const int lane = tid & 31;
    const int wave = tid >> 5;

    for (int i = tid; i < EXP * HDIM / 4; i += 256)
        ((float4*)sW3)[i] = ((const float4*)W3)[i];
    __syncthreads();

    #pragma unroll 1
    for (int t = 0; t < 8; ++t) {             // 8 tokens per wave, 64 per block
        const int m = blockIdx.x * 64 + wave * 8 + t;

        float acc[EXP];
        #pragma unroll
        for (int e = 0; e < EXP; ++e) acc[e] = 0.0f;

        const unsigned short* grow = g2 + (size_t)m * HDIM;
        for (int k = lane; k < HDIM; k += 32) {
            float g = bf2f(grow[k]);
            #pragma unroll
            for (int e = 0; e < EXP; ++e) acc[e] += g * sW3[e * HDIM + k];
        }
        #pragma unroll
        for (int e = 0; e < EXP; ++e)
            #pragma unroll
            for (int off = 16; off > 0; off >>= 1)
                acc[e] += __shfl_xor(acc[e], off, 32);

        float mx = -3.0e38f;
        #pragma unroll
        for (int e = 0; e < EXP; ++e) { acc[e] += b3[e]; mx = fmaxf(mx, acc[e]); }
        float s = 0.0f;
        #pragma unroll
        for (int e = 0; e < EXP; ++e) { acc[e] = expf(acc[e] - mx); s += acc[e]; }
        const float inv = 1.0f / s;
        if (lane < EXP) wout[(size_t)m * EXP + lane] = acc[lane] * inv;
    }
}

// ---------- launch ----------
extern "C" void kernel_launch(void* const* d_in, const int* in_sizes, int n_in,
                              void* d_out, int out_size, void* d_ws, size_t ws_size,
                              hipStream_t stream) {
    const float* x   = (const float*)d_in[0];
    const float* gW1 = (const float*)d_in[1];
    const float* gb1 = (const float*)d_in[2];
    const float* gW2 = (const float*)d_in[3];
    const float* gb2 = (const float*)d_in[4];
    const float* gW3 = (const float*)d_in[5];
    const float* gb3 = (const float*)d_in[6];
    const float* eW1 = (const float*)d_in[7];
    const float* eb1 = (const float*)d_in[8];
    const float* eW2 = (const float*)d_in[9];
    const float* eb2 = (const float*)d_in[10];

    float* out     = (float*)d_out;         // [TOK][DDIM]
    float* weights = out + OUT_ELEMS;       // [TOK][EXP]

    // Workspace: 8MB x_bf16 + 32MB bufA (g1 / expert h) + 32MB bufB (g2) = 72MB
    unsigned short* xbf  = (unsigned short*)d_ws;
    unsigned short* bufA = xbf  + (size_t)TOK * DDIM;
    unsigned short* bufB = bufA + (size_t)TOK * HDIM;

    // 1) x -> bf16
    f32_to_bf16_kernel<<<(TOK * DDIM / 8 + 255) / 256, 256, 0, stream>>>(x, xbf, TOK * DDIM / 8);

    // 2) g1 = gelu(x @ gW1^T + gb1)        [TOK,HDIM]
    gemm_bf16_kernel<0><<<dim3(HDIM / BN, TOK / BM), 256, 0, stream>>>(
        xbf, gW1, gb1, bufA, nullptr, TOK, HDIM, DDIM);

    // 3) g2 = gelu(g1 @ gW2^T + gb2)       [TOK,HDIM]
    gemm_bf16_kernel<0><<<dim3(HDIM / BN, TOK / BM), 256, 0, stream>>>(
        bufA, gW2, gb2, bufB, nullptr, TOK, HDIM, HDIM);

    // 4) weights = softmax(g2 @ gW3^T + gb3)  -> tail of d_out (gW3 staged in LDS)
    gate_softmax_kernel<<<TOK / 64, 256, EXP * HDIM * sizeof(float), stream>>>(
        bufB, gW3, gb3, weights);

    // 5) zero the weighted-sum accumulator
    hipMemsetAsync(out, 0, (size_t)OUT_ELEMS * sizeof(float), stream);

    // 6) experts: h_e = gelu(x @ eW1[e]^T + eb1[e]); out += w[:,e] * (h_e @ eW2[e]^T + eb2[e])
    for (int e = 0; e < EXP; ++e) {
        gemm_bf16_kernel<0><<<dim3(HDIM / BN, TOK / BM), 256, 0, stream>>>(
            xbf, eW1 + (size_t)e * HDIM * DDIM, eb1 + (size_t)e * HDIM,
            bufA, nullptr, TOK, HDIM, DDIM);
        gemm_bf16_kernel<1><<<dim3(DDIM / BN, TOK / BM), 256, 0, stream>>>(
            bufA, eW2 + (size_t)e * DDIM * HDIM, eb2 + (size_t)e * DDIM,
            out, weights + e, TOK, DDIM, HDIM);
    }
}
